// HD_70059506532414
// MI455X (gfx1250) — 8-bit matrix operand order repaired and checked by reading, compiled, not run on hardware
//
#include <hip/hip_runtime.h>
#include <stdint.h>

#define B_   8
#define C_   8
#define H_   512
#define W_   512
#define HW_  (H_ * W_)
#define BHW_ (B_ * HW_)

typedef int v8i __attribute__((ext_vector_type(8)));

__global__ void __launch_bounds__(256)
hd_argmax_kernel(const float* __restrict__ logits,
                 const float* __restrict__ target,
                 unsigned char* __restrict__ lx,
                 unsigned char* __restrict__ ty) {
    int p = blockIdx.x * blockDim.x + threadIdx.x;
    if (p >= BHW_) return;
    int b  = p >> 18;
    int hw = p & (HW_ - 1);
    const float* lp = logits + (size_t)b * C_ * HW_ + hw;
    const float* tp = target + (size_t)b * C_ * HW_ + hw;
    float lbest = lp[0]; int li = 0;
    float tbest = tp[0]; int ti = 0;
#pragma unroll
    for (int c = 1; c < C_; ++c) {
        float lv = lp[(size_t)c * HW_];
        float tv = tp[(size_t)c * HW_];
        if (lv > lbest) { lbest = lv; li = c; }
        if (tv > tbest) { tbest = tv; ti = c; }
    }
    lx[p] = (unsigned char)li;
    ty[p] = (unsigned char)ti;
}

__global__ void __launch_bounds__(256)
hd_rownorm_kernel(const unsigned char* __restrict__ lx,
                  const unsigned char* __restrict__ ty,
                  int* __restrict__ x2, int* __restrict__ y2) {
    int wid  = (blockIdx.x * blockDim.x + threadIdx.x) >> 5;
    int lane = threadIdx.x & 31;
    const unsigned char* map = (wid < B_ * H_) ? lx : ty;
    int* out = (wid < B_ * H_) ? x2 : y2;
    int row  = (wid < B_ * H_) ? wid : wid - B_ * H_;

    const uint4* rp = (const uint4*)(map + (size_t)row * W_);
    uint4 d = rp[lane];
    int s = 0;
    uint32_t words[4] = {d.x, d.y, d.z, d.w};
#pragma unroll
    for (int i = 0; i < 4; ++i) {
#pragma unroll
        for (int j = 0; j < 4; ++j) {
            int v = (int)((words[i] >> (8 * j)) & 0xFF);
            s += v * v;
        }
    }
#pragma unroll
    for (int m = 16; m >= 1; m >>= 1) s += __shfl_xor(s, m, 32);
    if (lane == 0) out[row] = s;
}

__global__ void __launch_bounds__(256)
hd_init_kernel(int* __restrict__ rowmin, int* __restrict__ colmin) {
    int i = blockIdx.x * blockDim.x + threadIdx.x;
    if (i < B_ * H_) { rowmin[i] = 0x7FFFFFFF; colmin[i] = 0x7FFFFFFF; }
}

__global__ void __launch_bounds__(256)
hd_gemm_kernel(const unsigned char* __restrict__ lx,
               const unsigned char* __restrict__ ty,
               const int* __restrict__ x2, const int* __restrict__ y2,
               int* __restrict__ rowmin, int* __restrict__ colmin) {
    int wid  = (blockIdx.x * blockDim.x + threadIdx.x) >> 5;
    int lane = threadIdx.x & 31;
    int b  = wid >> 10;
    int t  = wid & 1023;
    int tm = t >> 5;
    int tn = t & 31;
    int half = lane >> 4;
    int ln   = lane & 15;

    const uint32_t* arow =
        (const uint32_t*)(lx + (size_t)b * HW_ + (size_t)(tm * 16 + ln) * W_);
    const uint32_t* brow =
        (const uint32_t*)(ty + (size_t)b * HW_ + (size_t)(tn * 16 + ln) * W_);

    v8i acc = {0, 0, 0, 0, 0, 0, 0, 0};
#pragma unroll
    for (int ks = 0; ks < 8; ++ks) {
        int kb = ks * 64;
        v8i av, bv;
#pragma unroll
        for (int v = 0; v < 8; ++v) {
            int aoff = kb + ((v & 2) << 3) + ((v & 1) << 2) + (half << 3) +
                       ((v >> 2) << 5);
            int boff = kb + ((v & 2) << 3) + ((v & 1) << 2) + (half << 3) + ((v >> 2) << 5);
            av[v] = (int)arow[aoff >> 2];
            bv[v] = (int)brow[boff >> 2];
        }
        acc = __builtin_amdgcn_wmma_i32_16x16x64_iu8(
             false, av,  false, bv, acc,
             false,  false);
    }

    int y2n    = y2[b * H_ + tn * 16 + ln];
    int m_base = b * H_ + tm * 16 + half * 8;
    int cmin   = 0x7FFFFFFF;
#pragma unroll
    for (int r = 0; r < 8; ++r) {
        int x2m = x2[m_base + r];
        int d2  = x2m + y2n - 2 * acc[r];
        int rm  = d2;
        rm = min(rm, __shfl_xor(rm, 1, 32));
        rm = min(rm, __shfl_xor(rm, 2, 32));
        rm = min(rm, __shfl_xor(rm, 4, 32));
        rm = min(rm, __shfl_xor(rm, 8, 32));
        if (ln == 0) atomicMin(&rowmin[m_base + r], rm);
        cmin = min(cmin, d2);
    }
    cmin = min(cmin, __shfl_xor(cmin, 16, 32));
    if (half == 0) atomicMin(&colmin[b * H_ + tn * 16 + ln], cmin);
}

__global__ void __launch_bounds__(256)
hd_final_kernel(const int* __restrict__ rowmin, const int* __restrict__ colmin,
                float* __restrict__ out) {
    __shared__ int sm[256];
    int t = threadIdx.x;
    float sum = 0.0f;
    for (int b = 0; b < B_; ++b) {
        int v = -1;
        for (int i = t; i < 2 * H_; i += 256) {
            int x = (i < H_) ? rowmin[b * H_ + i] : colmin[b * H_ + i - H_];
            v = max(v, x);
        }
        sm[t] = v;
        __syncthreads();
        for (int s = 128; s > 0; s >>= 1) {
            if (t < s) sm[t] = max(sm[t], sm[t + s]);
            __syncthreads();
        }
        if (t == 0) sum += sqrtf((float)sm[0]);
        __syncthreads();
    }
    if (t == 0) out[0] = sum * (1.0f / (float)B_);
}

extern "C" void kernel_launch(void* const* d_in, const int* in_sizes, int n_in,
                              void* d_out, int out_size, void* d_ws,
                              size_t ws_size, hipStream_t stream) {
    const float* logits = (const float*)d_in[0];
    const float* target = (const float*)d_in[1];
    float* out = (float*)d_out;

    unsigned char* lx = (unsigned char*)d_ws;
    unsigned char* ty = lx + (size_t)BHW_;
    int* x2     = (int*)(ty + (size_t)BHW_);
    int* y2     = x2 + B_ * H_;
    int* rowmin = y2 + B_ * H_;
    int* colmin = rowmin + B_ * H_;

    hd_argmax_kernel<<<BHW_ / 256, 256, 0, stream>>>(logits, target, lx, ty);

    hd_rownorm_kernel<<<(2 * B_ * H_) / 8, 256, 0, stream>>>(lx, ty, x2, y2);

    hd_init_kernel<<<(B_ * H_ + 255) / 256, 256, 0, stream>>>(rowmin, colmin);

    hd_gemm_kernel<<<(B_ * 1024) / 8, 256, 0, stream>>>(lx, ty, x2, y2,
                                                        rowmin, colmin);

    hd_final_kernel<<<1, 256, 0, stream>>>(rowmin, colmin, out);
}
